// QuietAttention_47158740910631
// MI455X (gfx1250) — matrix operand order repaired and checked by reading, not compiled, not run on hardware
//
#include <hip/hip_runtime.h>
#include <hip/hip_bf16.h>

// ---------------------------------------------------------------------------
// QuietAttention on MI455X (gfx1250, wave32, WMMA).
//   scores = Q K^T / sqrt(D), mask==0 -> -1e9, sink column (-1e9) prepended
//   p = exp(s - max) / (1 + sum(exp(s - max)))     (softmax_one)
//   out = p @ [0; V]
// bf16 WMMA (v_wmma_f32_16x16x32_bf16) with f32 accumulate; softmax in f32.
// One workgroup (8 wave32s) per (batch, 16-query block); full 16x2048 score
// row lives in LDS (~198 KB of the WGP's 320 KB). Mask reads and p_attn
// writes use non-temporal cache hints (single-use, 268 MB each); Q/K/V stay
// resident in L2. PV uses double-buffered 128-key V stages in LDS.
// ---------------------------------------------------------------------------

typedef __attribute__((ext_vector_type(16))) __bf16 bf16x16;
typedef __attribute__((ext_vector_type(8)))  float  f32x8;

#define B_   16
#define S_   2048
#define D_   128
#define NW   8          // waves per workgroup (256 threads)
#define SPAD 2064       // padded LDS row stride (elements)
#define KT   128        // keys per V stage in phase 4

__device__ __forceinline__ f32x8 wmma_bf16(bf16x16 a, bf16x16 b, f32x8 c) {
  // (neg_a, A, neg_b, B, c_mod, C, reuse_a, reuse_b)
  return __builtin_amdgcn_wmma_f32_16x16x32_bf16(false, a, false, b,
                                                 (short)0, c, false, false);
}

__global__ void __launch_bounds__(256, 1)
quiet_attn_kernel(const float* __restrict__ q,
                  const float* __restrict__ k,
                  const float* __restrict__ v,
                  const int*   __restrict__ mask,
                  float* __restrict__ out,
                  float* __restrict__ pattn) {
  extern __shared__ char smem[];
  float*  scores = (float*)smem;                               // 16 x SPAD f32
  __bf16* pbuf   = (__bf16*)(smem + 16 * SPAD * 4);            // 16 x SPAD bf16
  float*  rowmax = (float*)(smem + 16 * SPAD * 4 + 16 * SPAD * 2);
  float*  sumrow = rowmax + 16;
  // Phase-4 V staging buffers alias the (dead) score buffer: 2 x 32 KB.
  __bf16* vbuf0 = (__bf16*)smem;
  __bf16* vbuf1 = (__bf16*)smem + (size_t)KT * D_;

  const int tid  = threadIdx.x;
  const int wv   = tid >> 5;        // wave id 0..7
  const int lane = tid & 31;
  const int half = lane >> 4;       // 0: lanes 0-15, 1: lanes 16-31
  const int ln   = lane & 15;

  const int b  = blockIdx.x / (S_ / 16);
  const int q0 = (blockIdx.x % (S_ / 16)) * 16;

  const float SCALE = 0.08838834764831845f; // 1/sqrt(128)

  // ---------------- Phase 1: scores = QK^T * scale, masked -> LDS ----------
  // A operand (Q tile 16x32 per K-chunk), ISA 16-bit A layout:
  //   row M = ln; elems 0..7 -> K = half*8+e; elems 8..15 -> K = 16+half*8+(e-8)
  bf16x16 aa[4];
  {
    const float* qrow = q + ((size_t)b * S_ + q0 + ln) * D_;
    #pragma unroll
    for (int c = 0; c < 4; ++c) {
      const float* p0 = qrow + c * 32 + half * 8;
      const float* p1 = qrow + c * 32 + 16 + half * 8;
      bf16x16 t;
      #pragma unroll
      for (int i = 0; i < 8; ++i) t[i] = (__bf16)p0[i];
      #pragma unroll
      for (int i = 0; i < 8; ++i) t[8 + i] = (__bf16)p1[i];
      aa[c] = t;
    }
  }

  for (int j = wv; j < S_ / 16; j += NW) {            // 16 key-tiles per wave
    const float* krow = k + ((size_t)b * S_ + j * 16 + ln) * D_;
    f32x8 acc = (f32x8)0.0f;
    #pragma unroll
    for (int c = 0; c < 4; ++c) {
      // B operand (K^T chunk 32x16): col N = ln, elem e<8 -> Kdim = half*8+e; e>=8 -> 16+half*8+(e-8)
      const float* kp = krow + c * 32;
      bf16x16 bb;
      #pragma unroll
      for (int i = 0; i < 16; ++i) bb[i] = (__bf16)kp[i < 8 ? half * 8 + i : 16 + half * 8 + (i - 8)];
      acc = wmma_bf16(aa[c], bb, acc);
    }
    // C/D layout: VGPR r -> row m = r + half*8, col n = ln
    const int* mrow = mask + ((size_t)b * S_ + q0) * S_ + (size_t)j * 16 + ln;
    #pragma unroll
    for (int r = 0; r < 8; ++r) {
      const int m = r + half * 8;
      const int mv = __builtin_nontemporal_load(&mrow[(size_t)m * S_]);
      const float s = mv ? acc[r] * SCALE : -1e9f;
      scores[m * SPAD + j * 16 + ln] = s;
    }
  }
  __syncthreads();

  // ---------------- Phase 2: row max, then e = exp(s-max) in place + sum ---
  const int rr  = tid >> 4;   // row 0..15 (16 threads per row)
  const int sub = tid & 15;
  float* srow = scores + rr * SPAD;

  float mx = -1e9f;           // sink score is exactly -1e9, folded in here
  for (int n = sub * 4; n < S_; n += 64) {
    const float4 s4 = *(const float4*)(srow + n);
    mx = fmaxf(mx, fmaxf(fmaxf(s4.x, s4.y), fmaxf(s4.z, s4.w)));
  }
  #pragma unroll
  for (int o = 8; o >= 1; o >>= 1) mx = fmaxf(mx, __shfl_xor(mx, o, 32));
  if (sub == 0) rowmax[rr] = mx;
  __syncthreads();

  const float rm = rowmax[rr];
  float sm = 0.0f;
  for (int n = sub * 4; n < S_; n += 64) {
    float4 s4 = *(const float4*)(srow + n);
    s4.x = expf(s4.x - rm); s4.y = expf(s4.y - rm);
    s4.z = expf(s4.z - rm); s4.w = expf(s4.w - rm);
    sm += (s4.x + s4.y) + (s4.z + s4.w);
    *(float4*)(srow + n) = s4;               // keep e for phase 3
  }
  #pragma unroll
  for (int o = 8; o >= 1; o >>= 1) sm += __shfl_xor(sm, o, 32);
  if (sub == 0) sumrow[rr] = sm;
  __syncthreads();

  // ---------------- Phase 3: p = e / (1 + e_sink + sum); emit p_attn -------
  const float esink = expf(-1e9f - rm);      // ~0 unless row fully masked
  const float inv   = 1.0f / (1.0f + esink + sumrow[rr]);
  float* prow = pattn + ((size_t)b * S_ + q0 + rr) * (size_t)(S_ + 1);
  if (sub == 0) __builtin_nontemporal_store(esink * inv, &prow[0]); // sink col
  for (int n = sub * 4; n < S_; n += 64) {
    const float4 e4 = *(const float4*)(srow + n);
    const float p0 = e4.x * inv, p1 = e4.y * inv;
    const float p2 = e4.z * inv, p3 = e4.w * inv;
    __builtin_nontemporal_store(p0, &prow[1 + n + 0]);
    __builtin_nontemporal_store(p1, &prow[1 + n + 1]);
    __builtin_nontemporal_store(p2, &prow[1 + n + 2]);
    __builtin_nontemporal_store(p3, &prow[1 + n + 3]);
    pbuf[rr * SPAD + n + 0] = (__bf16)p0;    // 4 consecutive bf16 -> b64 store
    pbuf[rr * SPAD + n + 1] = (__bf16)p1;
    pbuf[rr * SPAD + n + 2] = (__bf16)p2;
    pbuf[rr * SPAD + n + 3] = (__bf16)p3;
  }
  __syncthreads();   // also fences scores region before vbuf aliasing reuse

  // ---------------- Phase 4: out = P @ V, double-buffered V stages ---------
  // Stage kt: keys [kt*KT, kt*KT+KT) x all 128 d -> bf16 in vbuf[kt&1].
  // Each thread copies 64 contiguous floats (16x b128 loads, 8x b128 stores).
  f32x8 acc = (f32x8)0.0f;                   // wave wv owns d-cols wv*16..+15
  {
    const int key0 = tid >> 1;
    const int d0   = (tid & 1) * 64;
    {  // stage kt = 0
      const float* vp = v + ((size_t)b * S_ + key0) * D_ + d0;
      __bf16* dst = vbuf0 + key0 * D_ + d0;
      #pragma unroll
      for (int i = 0; i < 64; ++i) dst[i] = (__bf16)vp[i];
    }
    for (int kt = 0; kt < S_ / KT; ++kt) {
      __syncthreads();                       // stage kt visible; buffers safe
      if (kt + 1 < S_ / KT) {                // prefetch next stage
        const float* vp = v + ((size_t)b * S_ + (kt + 1) * KT + key0) * D_ + d0;
        __bf16* dst = ((kt + 1) & 1 ? vbuf1 : vbuf0) + key0 * D_ + d0;
        #pragma unroll
        for (int i = 0; i < 64; ++i) dst[i] = (__bf16)vp[i];
      }
      const __bf16* vb_base = (kt & 1) ? vbuf1 : vbuf0;
      #pragma unroll
      for (int c = 0; c < 4; ++c) {          // 4 x 32-key chunks per stage
        // A = P chunk 16x32 from LDS (same A layout; K-dim -> key index)
        const __bf16* pr = pbuf + ln * SPAD + kt * KT + c * 32;
        bf16x16 pa;
        #pragma unroll
        for (int i = 0; i < 8; ++i) pa[i] = pr[half * 8 + i];
        #pragma unroll
        for (int i = 0; i < 8; ++i) pa[8 + i] = pr[16 + half * 8 + i];
        // B = V chunk 32x16: col n = ln (d = wv*16+ln), elem e<8 -> key half*8+e; e>=8 -> 16+half*8+(e-8)
        bf16x16 vb;
        #pragma unroll
        for (int e = 0; e < 16; ++e)
          vb[e] = vb_base[(c * 32 + (e < 8 ? half * 8 + e : 16 + half * 8 + (e - 8))) * D_ + wv * 16 + ln];
        acc = wmma_bf16(pa, vb, acc);
      }
    }
  }
  #pragma unroll
  for (int r = 0; r < 8; ++r) {
    const int m = r + half * 8;
    out[((size_t)b * S_ + q0 + m) * D_ + wv * 16 + ln] = acc[r];
  }
}

extern "C" void kernel_launch(void* const* d_in, const int* in_sizes, int n_in,
                              void* d_out, int out_size, void* d_ws, size_t ws_size,
                              hipStream_t stream) {
  (void)in_sizes; (void)n_in; (void)out_size; (void)d_ws; (void)ws_size;
  const float* q    = (const float*)d_in[0];
  const float* k    = (const float*)d_in[1];
  const float* v    = (const float*)d_in[2];
  const int*   mask = (const int*)d_in[3];
  float* out   = (float*)d_out;
  float* pattn = out + (size_t)B_ * S_ * D_;   // outputs concatenated: out, p_attn

  const size_t lds = (size_t)16 * SPAD * 4     // f32 scores / V staging alias
                   + (size_t)16 * SPAD * 2     // bf16 P
                   + 128;                      // rowmax + sumrow
  (void)hipFuncSetAttribute((const void*)quiet_attn_kernel,
                            hipFuncAttributeMaxDynamicSharedMemorySize, (int)lds);
  dim3 grid(B_ * (S_ / 16));
  dim3 block(256);
  quiet_attn_kernel<<<grid, block, lds, stream>>>(q, k, v, mask, out, pattn);
}
